// SimpleHGN_7851200217931
// MI455X (gfx1250) — compile-verified, matrix operand order checked by reading, not run on hardware
//
#include <hip/hip_runtime.h>

typedef __attribute__((ext_vector_type(16))) __bf16 bf16x16;
typedef __attribute__((ext_vector_type(8)))  float  f32x8;
typedef unsigned short ushort_t;

#define NC   4096
#define ND   2048
#define NTOK 6144
#define HID  512
#define NH   8
#define DH   64

__device__ __forceinline__ ushort_t f2bf(float f) {
    unsigned int u = __float_as_uint(f);
    unsigned int r = u + 0x7FFFu + ((u >> 16) & 1u);
    return (ushort_t)(r >> 16);
}

// Load one WMMA 16xK=32 operand slice (A-row or B-column pattern):
// elements 0..7 -> K = hi*8 + 0..7 ; elements 8..15 -> K = hi*8 + 16..23
__device__ __forceinline__ bf16x16 ldAB(const ushort_t* p, int hi) {
    union U { uint4 u[2]; bf16x16 v; } x;
    const uint4* q = (const uint4*)(p + hi * 8);
    x.u[0] = q[0];
    x.u[1] = q[2];
    return x.v;
}

__device__ __forceinline__ f32x8 wmma_bf16(bf16x16 a, bf16x16 b, f32x8 c) {
    return __builtin_amdgcn_wmma_f32_16x16x32_bf16(false, a, false, b, (short)0, c, false, false);
}

// ---------------- pack / convert kernels ----------------

__global__ void cvt_x_kernel(const float* __restrict__ cf, const float* __restrict__ df,
                             ushort_t* __restrict__ xb) {
    size_t idx = (size_t)blockIdx.x * blockDim.x + threadIdx.x;
    if (idx >= (size_t)NTOK * HID) return;
    float v = (idx < (size_t)NC * HID) ? cf[idx] : df[idx - (size_t)NC * HID];
    xb[idx] = f2bf(v);
}

// src is [K][N] row-major f32 ; dst is [N][K] row-major bf16
__global__ void transpose_w_kernel(const float* __restrict__ src, ushort_t* __restrict__ dst,
                                   int K, int N) {
    size_t idx = (size_t)blockIdx.x * blockDim.x + threadIdx.x;
    if (idx >= (size_t)K * N) return;
    size_t k = idx / N, n = idx % N;
    dst[n * (size_t)K + k] = f2bf(src[idx]);
}

// adjT[j][i] = (cd_adj[i][j] != 0), u8, i in [0,NC), j in [0,ND)
__global__ void adjT_kernel(const int* __restrict__ adj, unsigned char* __restrict__ adjT) {
    size_t idx = (size_t)blockIdx.x * blockDim.x + threadIdx.x;
    if (idx >= (size_t)NC * ND) return;
    size_t i = idx / ND, j = idx % ND;
    adjT[j * (size_t)NC + i] = adj[idx] ? 1 : 0;
}

// ---------------- QKV projection GEMM (bf16 WMMA) ----------------
// One wave computes a 16(M) x 64(N) tile of x@W + b.  z selects Q/K/V.
// Q,K stored row-major bf16; V stored transposed Vt[col][token].
__global__ __launch_bounds__(32) void qkv_kernel(
    const ushort_t* __restrict__ xb,
    const ushort_t* __restrict__ WT0, const ushort_t* __restrict__ WT1, const ushort_t* __restrict__ WT2,
    const float* __restrict__ b0, const float* __restrict__ b1, const float* __restrict__ b2,
    ushort_t* __restrict__ Qb, ushort_t* __restrict__ Kb, ushort_t* __restrict__ Vt) {
    int lane = threadIdx.x, lan = lane & 15, hi = lane >> 4;
    int m0 = blockIdx.x * 16, n0 = blockIdx.y * 64, z = blockIdx.z;
    const ushort_t* WT = (z == 0) ? WT0 : (z == 1) ? WT1 : WT2;
    const float* bias = (z == 0) ? b0 : (z == 1) ? b1 : b2;

    f32x8 acc[4] = {};
    const ushort_t* arow = xb + (size_t)(m0 + lan) * HID;
    for (int kc = 0; kc < HID; kc += 32) {
        bf16x16 a = ldAB(arow + kc, hi);
#pragma unroll
        for (int t = 0; t < 4; ++t) {
            bf16x16 b = ldAB(WT + (size_t)(n0 + t * 16 + lan) * HID + kc, hi);
            acc[t] = wmma_bf16(a, b, acc[t]);
        }
    }
#pragma unroll
    for (int t = 0; t < 4; ++t) {
        int col = n0 + t * 16 + lan;
        float bb = bias[col];
#pragma unroll
        for (int r = 0; r < 8; ++r) {
            int row = m0 + hi * 8 + r;
            ushort_t v = f2bf(acc[t][r] + bb);
            if (z == 0)      Qb[(size_t)row * HID + col] = v;
            else if (z == 1) Kb[(size_t)row * HID + col] = v;
            else             Vt[(size_t)col * NTOK + row] = v;
        }
    }
}

// ---------------- streaming masked attention (flash style) ----------------
// One wave = (16 query tokens, 1 head). Streams opposite-side keys 32 at a time.
__global__ __launch_bounds__(32) void attn_kernel(
    const ushort_t* __restrict__ Qb, const ushort_t* __restrict__ Kb,
    const ushort_t* __restrict__ Vt, const int* __restrict__ cd_adj,
    const unsigned char* __restrict__ adjT, ushort_t* __restrict__ outb) {
    __shared__ __align__(16) ushort_t pbuf[16 * 32];
    int lane = threadIdx.x, lan = lane & 15, hi = lane >> 4;
    int q0 = blockIdx.x * 16, head = blockIdx.y;
    bool cside = (q0 < NC);
    int nk    = cside ? ND : NC;   // number of opposite-side keys
    int ktok0 = cside ? NC : 0;    // token offset of keys
    int iq0   = cside ? q0 : (q0 - NC);  // query index within its side

    const ushort_t* qrow = Qb + (size_t)(q0 + lan) * HID + head * DH;
    bf16x16 aq0 = ldAB(qrow, hi);
    bf16x16 aq1 = ldAB(qrow + 32, hi);

    float m[8], l[8];
    f32x8 acc[4] = {};
#pragma unroll
    for (int r = 0; r < 8; ++r) { m[r] = -1e30f; l[r] = 0.f; }

    for (int jb = 0; jb < nk; jb += 32) {
        f32x8 s[2];
#pragma unroll
        for (int st = 0; st < 2; ++st) {
            int j = jb + st * 16 + lan;
            const ushort_t* krow = Kb + (size_t)(ktok0 + j) * HID + head * DH;
            f32x8 z = {};
            z = wmma_bf16(aq0, ldAB(krow, hi), z);
            z = wmma_bf16(aq1, ldAB(krow + 32, hi), z);
            if (cside) {
                const unsigned char* mp = adjT + (size_t)j * NC + iq0 + hi * 8;
                uint2 mm = *(const uint2*)mp;
#pragma unroll
                for (int r = 0; r < 8; ++r) {
                    unsigned int b = ((r < 4 ? mm.x : mm.y) >> ((r & 3) * 8)) & 0xFFu;
                    z[r] = b ? z[r] * 0.125f : -1e9f;
                }
            } else {
                const int* mp = cd_adj + (size_t)j * ND + iq0 + hi * 8;
                uint4 a0 = *(const uint4*)mp;
                uint4 a1 = *(const uint4*)(mp + 4);
                unsigned int mv[8] = {a0.x, a0.y, a0.z, a0.w, a1.x, a1.y, a1.z, a1.w};
#pragma unroll
                for (int r = 0; r < 8; ++r) z[r] = mv[r] ? z[r] * 0.125f : -1e9f;
            }
            s[st] = z;
        }
        // online softmax update (row reductions stay within each 16-lane half)
        float p0[8], p1[8];
#pragma unroll
        for (int r = 0; r < 8; ++r) {
            float t = fmaxf(s[0][r], s[1][r]);
#pragma unroll
            for (int k = 1; k < 16; k <<= 1) t = fmaxf(t, __shfl_xor(t, k, 32));
            float mn = fmaxf(m[r], t);
            float alpha = __expf(m[r] - mn);
            float a0 = __expf(s[0][r] - mn);
            float a1 = __expf(s[1][r] - mn);
            float rs = a0 + a1;
#pragma unroll
            for (int k = 1; k < 16; k <<= 1) rs += __shfl_xor(rs, k, 32);
            l[r] = l[r] * alpha + rs;
            m[r] = mn;
            p0[r] = a0; p1[r] = a1;
#pragma unroll
            for (int dt = 0; dt < 4; ++dt) acc[dt][r] *= alpha;
        }
        // stage P into LDS (C-layout -> A-layout re-shuffle)
#pragma unroll
        for (int r = 0; r < 8; ++r) {
            int row = hi * 8 + r;
            pbuf[row * 32 + lan]      = f2bf(p0[r]);
            pbuf[row * 32 + 16 + lan] = f2bf(p1[r]);
        }
        __syncthreads();
        bf16x16 ap;
        {
            union U { uint4 u[2]; bf16x16 v; } x;
            const uint4* q = (const uint4*)&pbuf[lan * 32 + hi * 8];
            x.u[0] = q[0];
            x.u[1] = q[2];
            ap = x.v;
        }
#pragma unroll
        for (int dt = 0; dt < 4; ++dt) {
            const ushort_t* vrow = Vt + (size_t)(head * DH + dt * 16 + lan) * NTOK + ktok0 + jb;
            acc[dt] = wmma_bf16(ap, ldAB(vrow, hi), acc[dt]);
        }
        __syncthreads();
    }
#pragma unroll
    for (int r = 0; r < 8; ++r) l[r] = 1.f / l[r];
#pragma unroll
    for (int dt = 0; dt < 4; ++dt) {
        int col = head * DH + dt * 16 + lan;
#pragma unroll
        for (int r = 0; r < 8; ++r) {
            int row = q0 + hi * 8 + r;
            outb[(size_t)row * HID + col] = f2bf(acc[dt][r] * l[r]);
        }
    }
}

// ---------------- output GEMM: h = concat(x, out) @ Wu + bu ----------------
__global__ __launch_bounds__(32) void out_gemm_kernel(
    const ushort_t* __restrict__ xb, const ushort_t* __restrict__ outb,
    const ushort_t* __restrict__ WuT, const float* __restrict__ bu,
    float* __restrict__ hbuf) {
    int lane = threadIdx.x, lan = lane & 15, hi = lane >> 4;
    int m0 = blockIdx.x * 16, n0 = blockIdx.y * 64;
    f32x8 acc[4] = {};
    const ushort_t* xrow = xb + (size_t)(m0 + lan) * HID;
    const ushort_t* orow = outb + (size_t)(m0 + lan) * HID;
    for (int kc = 0; kc < 2 * HID; kc += 32) {
        bf16x16 a = (kc < HID) ? ldAB(xrow + kc, hi) : ldAB(orow + (kc - HID), hi);
#pragma unroll
        for (int t = 0; t < 4; ++t) {
            bf16x16 b = ldAB(WuT + (size_t)(n0 + t * 16 + lan) * (2 * HID) + kc, hi);
            acc[t] = wmma_bf16(a, b, acc[t]);
        }
    }
#pragma unroll
    for (int t = 0; t < 4; ++t) {
        int col = n0 + t * 16 + lan;
        float bb = bu[col];
#pragma unroll
        for (int r = 0; r < 8; ++r) {
            int row = m0 + hi * 8 + r;
            hbuf[(size_t)row * HID + col] = acc[t][r] + bb;
        }
    }
}

// ---------------- BatchNorm (training-mode batch stats) + ReLU ----------------
__global__ void bn_stats_kernel(const float* __restrict__ hbuf, float* __restrict__ stats) {
    int c = blockIdx.x * blockDim.x + threadIdx.x;
    if (c >= HID) return;
    float s = 0.f, s2 = 0.f;
    for (int i = 0; i < NTOK; ++i) {
        float v = hbuf[(size_t)i * HID + c];
        s += v; s2 += v * v;
    }
    float mu = s * (1.0f / NTOK);
    float var = s2 * (1.0f / NTOK) - mu * mu;
    stats[c] = mu;
    stats[HID + c] = rsqrtf(var + 1e-5f);
}

__global__ void bn_apply_kernel(const float* __restrict__ hbuf, const float* __restrict__ stats,
                                const float* __restrict__ gamma, const float* __restrict__ beta,
                                float* __restrict__ out) {
    size_t idx = (size_t)blockIdx.x * blockDim.x + threadIdx.x;
    if (idx >= (size_t)NTOK * HID) return;
    int c = (int)(idx & (HID - 1));
    float v = (hbuf[idx] - stats[c]) * stats[HID + c] * gamma[c] + beta[c];
    out[idx] = fmaxf(v, 0.f);
}

// ---------------- launch ----------------
static inline size_t alignup(size_t x) { return (x + 255) & ~(size_t)255; }

extern "C" void kernel_launch(void* const* d_in, const int* in_sizes, int n_in,
                              void* d_out, int out_size, void* d_ws, size_t ws_size,
                              hipStream_t stream) {
    const float* c_feat = (const float*)d_in[0];
    const float* d_feat = (const float*)d_in[1];
    const int*   cd_adj = (const int*)d_in[2];
    const float* Wq = (const float*)d_in[3];  const float* bq = (const float*)d_in[4];
    const float* Wk = (const float*)d_in[5];  const float* bk = (const float*)d_in[6];
    const float* Wv = (const float*)d_in[7];  const float* bv = (const float*)d_in[8];
    const float* Wu = (const float*)d_in[9];  const float* bu = (const float*)d_in[10];
    const float* gamma = (const float*)d_in[11];
    const float* beta  = (const float*)d_in[12];
    float* out = (float*)d_out;

    char* w = (char*)d_ws;
    size_t off = 0;
    ushort_t* xb  = (ushort_t*)(w + off); off += alignup((size_t)NTOK * HID * 2);
    ushort_t* WqT = (ushort_t*)(w + off); off += alignup((size_t)HID * HID * 2);
    ushort_t* WkT = (ushort_t*)(w + off); off += alignup((size_t)HID * HID * 2);
    ushort_t* WvT = (ushort_t*)(w + off); off += alignup((size_t)HID * HID * 2);
    ushort_t* WuT = (ushort_t*)(w + off); off += alignup((size_t)2 * HID * HID * 2);
    ushort_t* Qb  = (ushort_t*)(w + off); off += alignup((size_t)NTOK * HID * 2);
    ushort_t* Kb  = (ushort_t*)(w + off); off += alignup((size_t)NTOK * HID * 2);
    ushort_t* Vt  = (ushort_t*)(w + off); off += alignup((size_t)NTOK * HID * 2);
    ushort_t* outb = (ushort_t*)(w + off); off += alignup((size_t)NTOK * HID * 2);
    unsigned char* adjT = (unsigned char*)(w + off); off += alignup((size_t)NC * ND);
    float* hbuf  = (float*)(w + off); off += alignup((size_t)NTOK * HID * 4);
    float* stats = (float*)(w + off); off += alignup((size_t)2 * HID * 4);

    // pack / convert
    cvt_x_kernel<<<(NTOK * HID + 255) / 256, 256, 0, stream>>>(c_feat, d_feat, xb);
    transpose_w_kernel<<<(HID * HID + 255) / 256, 256, 0, stream>>>(Wq, WqT, HID, HID);
    transpose_w_kernel<<<(HID * HID + 255) / 256, 256, 0, stream>>>(Wk, WkT, HID, HID);
    transpose_w_kernel<<<(HID * HID + 255) / 256, 256, 0, stream>>>(Wv, WvT, HID, HID);
    transpose_w_kernel<<<(2 * HID * HID + 255) / 256, 256, 0, stream>>>(Wu, WuT, 2 * HID, HID);
    adjT_kernel<<<(NC * ND + 255) / 256, 256, 0, stream>>>(cd_adj, adjT);

    // QKV projection
    qkv_kernel<<<dim3(NTOK / 16, HID / 64, 3), 32, 0, stream>>>(
        xb, WqT, WkT, WvT, bq, bk, bv, Qb, Kb, Vt);

    // streaming masked attention
    attn_kernel<<<dim3(NTOK / 16, NH), 32, 0, stream>>>(Qb, Kb, Vt, cd_adj, adjT, outb);

    // output projection
    out_gemm_kernel<<<dim3(NTOK / 16, HID / 64), 32, 0, stream>>>(xb, outb, WuT, bu, hbuf);

    // batchnorm + relu
    bn_stats_kernel<<<(HID + 255) / 256, 256, 0, stream>>>(hbuf, stats);
    bn_apply_kernel<<<(NTOK * HID + 255) / 256, 256, 0, stream>>>(hbuf, stats, gamma, beta, out);
}
